// Model_29944511987740
// MI455X (gfx1250) — matrix operand order repaired and checked by reading, not compiled, not run on hardware
//
#include <hip/hip_runtime.h>
#include <math.h>

// ---------------------------------------------------------------------------
// Problem constants (from reference)
// ---------------------------------------------------------------------------
#define D_IN  1024
#define SIZE  64
#define NN    (SIZE * SIZE)   // 4096
#define BATCH 8192
#define CHUNK 2048            // batch rows per logits chunk (32MB f32 scratch)

// ---------------------------------------------------------------------------
// CDNA5 WMMA types
// ---------------------------------------------------------------------------
typedef __attribute__((ext_vector_type(16))) __bf16 v16bf;
typedef __attribute__((ext_vector_type(8)))  __bf16 v8bf;
typedef __attribute__((ext_vector_type(4)))  __bf16 v4bf;
typedef __attribute__((ext_vector_type(8)))  float  v8f;

union ABf { v16bf v; v8bf h[2]; };

__device__ __forceinline__ v8f wmma_bf16(v16bf a, v16bf b, v8f c) {
    return __builtin_amdgcn_wmma_f32_16x16x32_bf16(
        /*neg_a=*/false, a, /*neg_b=*/false, b,
        /*c_mod=*/(short)0, c, /*reuse_a=*/false, /*reuse_b=*/false);
}

__device__ __forceinline__ float wave_max(float v) {
#pragma unroll
    for (int off = 16; off > 0; off >>= 1)
        v = fmaxf(v, __shfl_xor(v, off, 32));
    return v;
}
__device__ __forceinline__ float wave_sum(float v) {
#pragma unroll
    for (int off = 16; off > 0; off >>= 1)
        v += __shfl_xor(v, off, 32);
    return v;
}

// ---------------------------------------------------------------------------
// f32 -> bf16 elementwise convert (vectorized x4)
// ---------------------------------------------------------------------------
__global__ void cvt_f32_bf16(const float* __restrict__ in,
                             __bf16* __restrict__ out, int n4) {
    int i = blockIdx.x * blockDim.x + threadIdx.x;
    if (i >= n4) return;
    float4 f = reinterpret_cast<const float4*>(in)[i];
    v4bf o;
    o.x = (__bf16)f.x; o.y = (__bf16)f.y; o.z = (__bf16)f.z; o.w = (__bf16)f.w;
    reinterpret_cast<v4bf*>(out)[i] = o;
}

// ---------------------------------------------------------------------------
// W[K,N] f32 (row-major [in,out]) -> Wt[N,K] bf16 (tiled LDS transpose).
// Makes every WMMA B-fragment one contiguous 32B load.
// ---------------------------------------------------------------------------
__global__ void transpose_cvt(const float* __restrict__ W,
                              __bf16* __restrict__ Wt, int K, int N) {
    __shared__ float tile[32][33];
    const int nBase = blockIdx.x * 32;
    const int kBase = blockIdx.y * 32;
    const int tx = threadIdx.x, ty = threadIdx.y;
#pragma unroll
    for (int j = 0; j < 32; j += 8)
        tile[ty + j][tx] = W[(size_t)(kBase + ty + j) * N + nBase + tx];
    __syncthreads();
#pragma unroll
    for (int j = 0; j < 32; j += 8)
        Wt[(size_t)(nBase + ty + j) * K + kBase + tx] = (__bf16)tile[tx][ty + j];
}

// ---------------------------------------------------------------------------
// Software-pipelined 32x64 per-wave GEMM core (8 accumulators).
// Double-buffered fragments: loads for stage k+32 are issued before the
// 8 WMMAs of stage k, so s_wait_loadcnt covers a full stage of latency.
// Fragment addressing (contiguous k order, same for A and B; not the ISA table order):
//   A lane(l): row = mBase + (l&15);  elems 0-15 contiguous at k+16*(l>>4)
//   B lane(l): col = nBase + (l&15);  elems 0-15 contiguous at k+16*(l>>4)
//   C vgpr r : row = mBase + r + 8*(l>>4), col = nBase + (l&15)
// ---------------------------------------------------------------------------
__device__ __forceinline__ void gemm_core_32x64(const __bf16* __restrict__ aP0,
                                                const __bf16* __restrict__ bP,
                                                int K, v8f (&acc)[2][4]) {
    const __bf16* aP1 = aP0 + (size_t)16 * K;

    ABf a0[2], a1[2];
    v16bf b0[4], b1[4];

    auto loadA = [&](ABf (&a)[2], int k) {
        a[0].h[0] = *reinterpret_cast<const v8bf*>(aP0 + k);
        a[0].h[1] = *reinterpret_cast<const v8bf*>(aP0 + k + 8);
        a[1].h[0] = *reinterpret_cast<const v8bf*>(aP1 + k);
        a[1].h[1] = *reinterpret_cast<const v8bf*>(aP1 + k + 8);
    };
    auto loadB = [&](v16bf (&b)[4], int k) {
#pragma unroll
        for (int t = 0; t < 4; ++t)
            b[t] = *reinterpret_cast<const v16bf*>(bP + (size_t)(t * 16) * K + k);
    };
    auto mma = [&](ABf (&a)[2], v16bf (&b)[4]) {
#pragma unroll
        for (int rb = 0; rb < 2; ++rb)
#pragma unroll
            for (int t = 0; t < 4; ++t)
                acc[rb][t] = wmma_bf16(a[rb].v, b[t], acc[rb][t]);
    };

    loadA(a0, 0); loadB(b0, 0);
    int k = 0;
#pragma unroll 1
    for (; k + 64 < K; k += 64) {
        loadA(a1, k + 32); loadB(b1, k + 32);
        mma(a0, b0);
        loadA(a0, k + 64); loadB(b0, k + 64);
        mma(a1, b1);
    }
    loadA(a1, K - 32); loadB(b1, K - 32);
    mma(a0, b0);
    mma(a1, b1);
}

// ---------------------------------------------------------------------------
// C[M,N] = relu(A[M,K] @ Bt[N,K]^T + bias), bf16 in / bf16 out.
// ---------------------------------------------------------------------------
__global__ __launch_bounds__(256)
void gemm_relu_bf16(const __bf16* __restrict__ A, const __bf16* __restrict__ Bt,
                    const float* __restrict__ bias, __bf16* __restrict__ C,
                    int N, int K) {
    const int lane = threadIdx.x & 31;
    const int wave = (blockIdx.x * blockDim.x + threadIdx.x) >> 5;
    const int tilesN = N >> 6;
    const int mBase = (wave / tilesN) << 5;
    const int nBase = (wave % tilesN) << 6;
    const int lr = lane & 15, hi = lane >> 4;

    const __bf16* aP0 = A  + (size_t)(mBase + lr) * K + hi * 16;
    const __bf16* bP  = Bt + (size_t)(nBase + lr) * K + hi * 16;

    v8f acc[2][4] = {};
    gemm_core_32x64(aP0, bP, K, acc);

#pragma unroll
    for (int rb = 0; rb < 2; ++rb)
#pragma unroll
        for (int t = 0; t < 4; ++t) {
            const int col = nBase + 16 * t + lr;
            const float bv = bias[col];
#pragma unroll
            for (int r = 0; r < 8; ++r) {
                const int row = mBase + rb * 16 + r + hi * 8;
                C[(size_t)row * N + col] = (__bf16)fmaxf(acc[rb][t][r] + bv, 0.0f);
            }
        }
}

// ---------------------------------------------------------------------------
// Logits chunk: C[M,N]=A@Bt^T+bias, f32 out (no activation).  A is offset to
// the chunk's first row by the host; C rows are chunk-relative.
// ---------------------------------------------------------------------------
__global__ __launch_bounds__(256)
void gemm_bias_f32(const __bf16* __restrict__ A, const __bf16* __restrict__ Bt,
                   const float* __restrict__ bias, float* __restrict__ C,
                   int N, int K) {
    const int lane = threadIdx.x & 31;
    const int wave = (blockIdx.x * blockDim.x + threadIdx.x) >> 5;
    const int tilesN = N >> 6;
    const int mBase = (wave / tilesN) << 5;
    const int nBase = (wave % tilesN) << 6;
    const int lr = lane & 15, hi = lane >> 4;

    const __bf16* aP0 = A  + (size_t)(mBase + lr) * K + hi * 16;
    const __bf16* bP  = Bt + (size_t)(nBase + lr) * K + hi * 16;

    v8f acc[2][4] = {};
    gemm_core_32x64(aP0, bP, K, acc);

#pragma unroll
    for (int rb = 0; rb < 2; ++rb)
#pragma unroll
        for (int t = 0; t < 4; ++t) {
            const int col = nBase + 16 * t + lr;
            const float bv = bias[col];
#pragma unroll
            for (int r = 0; r < 8; ++r) {
                const int row = mBase + rb * 16 + r + hi * 8;
                C[(size_t)row * N + col] = acc[rb][t][r] + bv;
            }
        }
}

// ---------------------------------------------------------------------------
// d2 = softmax(h_b @ W2b^T + b2b) over N=64.  One wave per 16 batch rows;
// 16x64 logit tile through LDS (stride 65), per-row wave-level softmax.
// ---------------------------------------------------------------------------
__global__ __launch_bounds__(256)
void gemm_softmax_d2(const __bf16* __restrict__ A, const __bf16* __restrict__ Bt,
                     const float* __restrict__ bias, float* __restrict__ d2,
                     int K) {
    __shared__ float sD[8 * 16 * 65];
    const int lane = threadIdx.x & 31;
    const int w    = threadIdx.x >> 5;
    const int mBase = (blockIdx.x * 8 + w) << 4;
    const int lr = lane & 15, hi = lane >> 4;

    const __bf16* aP = A  + (size_t)(mBase + lr) * K + hi * 16;
    const __bf16* bP = Bt + (size_t)lr * K + hi * 16;

    v8f acc[4] = {};
#pragma unroll 1
    for (int k0 = 0; k0 < K; k0 += 32) {
        ABf a;
        a.h[0] = *reinterpret_cast<const v8bf*>(aP + k0);
        a.h[1] = *reinterpret_cast<const v8bf*>(aP + k0 + 8);
#pragma unroll
        for (int t = 0; t < 4; ++t) {
            v16bf b = *reinterpret_cast<const v16bf*>(bP + (size_t)(t * 16) * K + k0);
            acc[t] = wmma_bf16(a.v, b, acc[t]);
        }
    }
    float* sW = sD + w * 16 * 65;
#pragma unroll
    for (int t = 0; t < 4; ++t) {
        const float bv = bias[16 * t + lr];
#pragma unroll
        for (int r = 0; r < 8; ++r)
            sW[(r + 8 * hi) * 65 + 16 * t + lr] = acc[t][r] + bv;
    }
#pragma unroll 1
    for (int row = 0; row < 16; ++row) {
        const float v0 = sW[row * 65 + lane];
        const float v1 = sW[row * 65 + lane + 32];
        const float m  = wave_max(fmaxf(v0, v1));
        const float e0 = __expf(v0 - m), e1 = __expf(v1 - m);
        const float inv = 1.0f / wave_sum(e0 + e1);
        float* o = d2 + (size_t)(mBase + row) * SIZE;
        o[lane]      = e0 * inv;
        o[lane + 32] = e1 * inv;
    }
}

// ---------------------------------------------------------------------------
// Per-row softmax over 4096 logits + min/max join against d2.
// One block per batch row: stage exp() row in LDS, block-reduce max/sum,
// then out[b,y] = max_x min(p[x*64+y], d2[b,x]).
// ---------------------------------------------------------------------------
__global__ __launch_bounds__(256)
void softmax_join(const float* __restrict__ logits,  // [CHUNK, 4096]
                  const float* __restrict__ d2,      // [BATCH, 64]
                  float* __restrict__ out,           // [BATCH, 64]
                  int rowBase) {
    extern __shared__ float s[];
    float* sE   = s;            // 4096 floats
    float* sRed = s + NN;       // 8 floats
    const int t = threadIdx.x;
    const int lane = t & 31, w = t >> 5;
    const float* lp = logits + (size_t)blockIdx.x * NN;

    // pass 1: load row (16 elems/thread as float4) + max
    float4 v[4];
    float mx = -INFINITY;
#pragma unroll
    for (int j = 0; j < 4; ++j) {
        v[j] = reinterpret_cast<const float4*>(lp)[j * 256 + t];
        mx = fmaxf(mx, fmaxf(fmaxf(v[j].x, v[j].y), fmaxf(v[j].z, v[j].w)));
    }
    mx = wave_max(mx);
    if (lane == 0) sRed[w] = mx;
    __syncthreads();
    float bm = sRed[0];
#pragma unroll
    for (int i = 1; i < 8; ++i) bm = fmaxf(bm, sRed[i]);

    // pass 2: exp -> LDS, sum
    float sum = 0.0f;
#pragma unroll
    for (int j = 0; j < 4; ++j) {
        float4 e;
        e.x = __expf(v[j].x - bm); e.y = __expf(v[j].y - bm);
        e.z = __expf(v[j].z - bm); e.w = __expf(v[j].w - bm);
        sum += (e.x + e.y) + (e.z + e.w);
        reinterpret_cast<float4*>(sE)[j * 256 + t] = e;
    }
    sum = wave_sum(sum);
    __syncthreads();                 // sRed consumed & sE fully written
    if (lane == 0) sRed[w] = sum;
    __syncthreads();
    float bs = 0.0f;
#pragma unroll
    for (int i = 0; i < 8; ++i) bs += sRed[i];
    const float inv = 1.0f / bs;

    // pass 3: join (64 output columns)
    if (t < SIZE) {
        const int gRow = rowBase + blockIdx.x;
        const float* d2r = d2 + (size_t)gRow * SIZE;
        float m = 0.0f;
#pragma unroll 1
        for (int x = 0; x < SIZE; ++x)
            m = fmaxf(m, fminf(sE[x * SIZE + t] * inv, d2r[x]));
        out[(size_t)gRow * SIZE + t] = m;
    }
}

// ---------------------------------------------------------------------------
// Host-side orchestration
// ---------------------------------------------------------------------------
extern "C" void kernel_launch(void* const* d_in, const int* in_sizes, int n_in,
                              void* d_out, int out_size, void* d_ws, size_t ws_size,
                              hipStream_t stream) {
    (void)in_sizes; (void)n_in; (void)out_size; (void)ws_size;
    const float* x   = (const float*)d_in[0];
    const float* W1a = (const float*)d_in[1];
    const float* b1a = (const float*)d_in[2];
    const float* W2a = (const float*)d_in[3];
    const float* b2a = (const float*)d_in[4];
    const float* W1b = (const float*)d_in[5];
    const float* b1b = (const float*)d_in[6];
    const float* W2b = (const float*)d_in[7];
    const float* b2b = (const float*)d_in[8];
    float* out = (float*)d_out;

    // workspace layout (~96 MB total)
    char* ws = (char*)d_ws;
    size_t off = 0;
    auto alloc = [&](size_t bytes) -> char* {
        char* p = ws + off;
        off += (bytes + 255) & ~(size_t)255;
        return p;
    };
    __bf16* xb     = (__bf16*)alloc((size_t)BATCH * D_IN * 2);
    __bf16* w1at   = (__bf16*)alloc((size_t)D_IN * D_IN * 2);
    __bf16* w1bt   = (__bf16*)alloc((size_t)D_IN * D_IN * 2);
    __bf16* w2at   = (__bf16*)alloc((size_t)D_IN * NN   * 2);
    __bf16* w2bt   = (__bf16*)alloc((size_t)D_IN * SIZE * 2);
    __bf16* ha     = (__bf16*)alloc((size_t)BATCH * D_IN * 2);
    __bf16* hb     = (__bf16*)alloc((size_t)BATCH * D_IN * 2);
    float*  d2     = (float*) alloc((size_t)BATCH * SIZE * 4);
    float*  logits = (float*) alloc((size_t)CHUNK * NN   * 4);

    // 1) precision/layout transforms
    {
        const int n4 = BATCH * D_IN / 4;
        cvt_f32_bf16<<<(n4 + 255) / 256, 256, 0, stream>>>(x, xb, n4);
        dim3 tb(32, 8);
        transpose_cvt<<<dim3(D_IN / 32, D_IN / 32), tb, 0, stream>>>(W1a, w1at, D_IN, D_IN);
        transpose_cvt<<<dim3(D_IN / 32, D_IN / 32), tb, 0, stream>>>(W1b, w1bt, D_IN, D_IN);
        transpose_cvt<<<dim3(NN   / 32, D_IN / 32), tb, 0, stream>>>(W2a, w2at, D_IN, NN);
        transpose_cvt<<<dim3(SIZE / 32, D_IN / 32), tb, 0, stream>>>(W2b, w2bt, D_IN, SIZE);
    }

    // 2) layer-1 GEMMs (M=8192, N=1024, K=1024), relu, bf16 out
    {
        const int waves = (BATCH / 32) * (D_IN / 64);   // 4096 waves
        gemm_relu_bf16<<<waves / 8, 256, 0, stream>>>(xb, w1at, b1a, ha, D_IN, D_IN);
        gemm_relu_bf16<<<waves / 8, 256, 0, stream>>>(xb, w1bt, b1b, hb, D_IN, D_IN);
    }

    // 3) d2 = softmax(hb @ W2b + b2b)   (M=8192, N=64, K=1024)
    gemm_softmax_d2<<<BATCH / 16 / 8, 256, 0, stream>>>(hb, w2bt, b2b, d2, D_IN);

    // 4) chunked logits_a GEMM + softmax/join (logits buffer reused per chunk)
    {
        const int gblocks = (CHUNK / 32) * (NN / 64) / 8;       // 512
        const size_t shmem = (size_t)(NN + 8) * sizeof(float);  // 16.4 KB
        for (int c = 0; c < BATCH; c += CHUNK) {
            gemm_bias_f32<<<gblocks, 256, 0, stream>>>(
                ha + (size_t)c * D_IN, w2at, b2a, logits, NN, D_IN);
            softmax_join<<<CHUNK, 256, shmem, stream>>>(logits, d2, out, c);
        }
    }
}
